// NodeTrans_38482906972443
// MI455X (gfx1250) — matrix operand order repaired and checked by reading, not compiled, not run on hardware
//
#include <hip/hip_runtime.h>
#include <hip/hip_bf16.h>

typedef __attribute__((ext_vector_type(16))) __bf16 v16bf;
typedef __attribute__((ext_vector_type(8)))  float  v8f;
typedef __attribute__((ext_vector_type(4)))  float  v4f;   // native vector: valid for nontemporal builtins

// ---------------------------------------------------------------------------
// 1) flag setup
// ---------------------------------------------------------------------------
__global__ void k_init_flags(int* flags, int n) {
    int i = blockIdx.x * blockDim.x + threadIdx.x;
    if (i < n) flags[i] = 0;
}

__global__ void k_mark_flags(const int* __restrict__ node_id, int* flags, int b) {
    int i = blockIdx.x * blockDim.x + threadIdx.x;
    if (i < b) flags[node_id[i]] = 1;
}

// ---------------------------------------------------------------------------
// 2) pass-through copy of non-updated rows (v4f, non-temporal: streamed once)
// ---------------------------------------------------------------------------
__device__ __forceinline__ void copy_row_nt(const v4f* __restrict__ s,
                                            v4f* __restrict__ d, int n4, int t, int tb) {
    for (int i = t; i < n4; i += tb) {
        v4f v = __builtin_nontemporal_load(s + i);
        __builtin_nontemporal_store(v, d + i);
    }
}

__global__ void k_copy_rows(const float* __restrict__ feat, const float* __restrict__ img,
                            const float* __restrict__ kg,   const float* __restrict__ allm,
                            float* __restrict__ ofeat, float* __restrict__ oimg,
                            float* __restrict__ okg,   float* __restrict__ oall,
                            const int* __restrict__ flags, int H, int NI, int NK) {
    unsigned row = blockIdx.x;
    if (flags[row] != 0) return;            // updated rows written by compute kernels
    int t = threadIdx.x, tb = blockDim.x;
    int NA = NI + NK;
    copy_row_nt((const v4f*)(feat + (unsigned)(row * H)),  (v4f*)(ofeat + (unsigned)(row * H)),  H  >> 2, t, tb);
    copy_row_nt((const v4f*)(img  + (unsigned)(row * NI)), (v4f*)(oimg  + (unsigned)(row * NI)), NI >> 2, t, tb);
    copy_row_nt((const v4f*)(kg   + (unsigned)(row * NK)), (v4f*)(okg   + (unsigned)(row * NK)), NK >> 2, t, tb);
    copy_row_nt((const v4f*)(allm + (unsigned)(row * NA)), (v4f*)(oall  + (unsigned)(row * NA)), NA >> 2, t, tb);
}

// ---------------------------------------------------------------------------
// 3) pack all_node = concat(img_feat, kg_feat) into WMMA bf16 B-fragment layout
//    fragment element index = ((s*NT + n0)*32 + lane)*16 + e
//    K = s*32 + (lane>=16 ? 16 : 0) + e ; N = n0*16 + (lane&15)
// ---------------------------------------------------------------------------
__global__ void k_build_b(const float* __restrict__ img_feat, const float* __restrict__ kg_feat,
                          __bf16* __restrict__ bws, int H, int NI, int NA) {
    unsigned tid = blockIdx.x * blockDim.x + threadIdx.x;
    unsigned total = (unsigned)NA * (unsigned)H;
    if (tid >= total) return;
    int e    = (int)(tid & 15);
    int lane = (int)((tid >> 4) & 31);
    unsigned blk = tid >> 9;
    int NT = H >> 4;
    int n0 = (int)(blk % NT);
    int s  = (int)(blk / NT);
    int K  = s * 32 + ((lane & 16) ? 16 : 0) + e;
    int n  = n0 * 16 + (lane & 15);
    float v = (K < NI) ? img_feat[(unsigned)(K * H) + n]
                       : kg_feat[(unsigned)((K - NI) * H) + n];
    bws[tid] = (__bf16)v;
}

// ---------------------------------------------------------------------------
// 4) gate MLP: sigmoid(relu(h@w1 + b1)@w2 + b2)  — one 128-thread block per row
// ---------------------------------------------------------------------------
__global__ void __launch_bounds__(128)
k_gate(const float* __restrict__ feat, const int* __restrict__ node_id,
       const float* __restrict__ w1, const float* __restrict__ b1,
       const float* __restrict__ w2, const float* __restrict__ b2,
       float* __restrict__ gate, int H, int MIDc) {
    __shared__ float h[256];
    __shared__ float red[128];
    int b = blockIdx.x, t = threadIdx.x;
    unsigned row = (unsigned)node_id[b];
    for (int i = t; i < H; i += MIDc) h[i] = feat[row * H + i];
    __syncthreads();
    float acc = b1[t];
    #pragma unroll 4
    for (int k = 0; k < H; ++k) acc = fmaf(h[k], w1[(unsigned)(k * MIDc) + t], acc);
    float m = fmaxf(acc, 0.0f);
    red[t] = m * w2[t];
    __syncthreads();
    for (int s = MIDc >> 1; s > 0; s >>= 1) {
        if (t < s) red[t] += red[t + s];
        __syncthreads();
    }
    if (t == 0) {
        float z = red[0] + b2[0];
        gate[b] = 1.0f / (1.0f + __expf(-z));
    }
}

// ---------------------------------------------------------------------------
// 5) blend + joint softmax — one 256-thread block per row, values in registers
// ---------------------------------------------------------------------------
__global__ void __launch_bounds__(256)
k_blend_softmax(const float* __restrict__ img, const float* __restrict__ kg,
                const int* __restrict__ node_id, const int* __restrict__ pre_id,
                const float* __restrict__ gate,
                float* __restrict__ oimg, float* __restrict__ okg, float* __restrict__ oall,
                int NI, int NK) {
    int b = blockIdx.x, t = threadIdx.x;
    int NA = NI + NK;
    unsigned row  = (unsigned)node_id[b];
    unsigned prow = (unsigned)pre_id[b];
    float g = gate[b], omg = 1.0f - g;

    float v[16];                       // NA/256 <= 16
    float lmax = -3.402823466e38f;
    int cnt = 0;
    for (int i = t; i < NA; i += 256) {
        float c, p;
        if (i < NI) { c = img[row * NI + i];       p = img[prow * NI + i]; }
        else        { int j = i - NI; c = kg[row * NK + j]; p = kg[prow * NK + j]; }
        float nv = c * g + omg * p;
        if (i < NI) __builtin_nontemporal_store(nv, &oimg[row * NI + i]);
        else        __builtin_nontemporal_store(nv, &okg[row * NK + (i - NI)]);
        v[cnt++] = nv;
        lmax = fmaxf(lmax, nv);
    }
    __shared__ float red[256];
    red[t] = lmax;
    __syncthreads();
    for (int s = 128; s > 0; s >>= 1) { if (t < s) red[t] = fmaxf(red[t], red[t + s]); __syncthreads(); }
    float mx = red[0];
    __syncthreads();
    float lsum = 0.0f;
    cnt = 0;
    for (int i = t; i < NA; i += 256) { float e = __expf(v[cnt] - mx); v[cnt] = e; lsum += e; ++cnt; }
    red[t] = lsum;
    __syncthreads();
    for (int s = 128; s > 0; s >>= 1) { if (t < s) red[t] += red[t + s]; __syncthreads(); }
    float inv = 1.0f / red[0];
    cnt = 0;
    for (int i = t; i < NA; i += 256) oall[row * NA + i] = v[cnt++] * inv;  // regular: re-read by WMMA kernel
}

// ---------------------------------------------------------------------------
// 6) attention matmul: new_feat[B x H] = new_all[B x NA] @ all_node[NA x H]
//    bf16 WMMA, f32 accumulate. 1 block = 16 batch rows x 256 cols,
//    4 waves x 4 N-tiles. Software-pipelined:
//      - B fragments for step s+1 live across the iteration (loads issued
//        before step s's WMMAs, waited at step s+1) — cross-iteration
//        liveness prevents the register allocator from re-serializing them.
//      - A tile s+1: global load issued at top of step, ds_store AFTER the
//        WMMAs (ping-pong buffer, still before the next barrier).
// ---------------------------------------------------------------------------
__global__ void __launch_bounds__(128)
k_attn_wmma(const float* __restrict__ newall, const __bf16* __restrict__ bws,
            const int* __restrict__ node_id, float* __restrict__ ofeat,
            int B, int H, int NA) {
    __shared__ float lds_a[2][16 * 36];  // 16x32 f32 tile, row stride 36 (aligned, conflict-free)
    __shared__ int   nid[16];

    int t = threadIdx.x;
    int bbase = blockIdx.x * 16;
    if (t < 16) {
        int b = bbase + t;
        if (b >= B) b = B - 1;           // tail: duplicate last row (same value rewritten)
        nid[t] = node_id[b];
    }
    __syncthreads();

    int lane = t & 31;
    int wave = t >> 5;
    int m    = lane & 15;
    int koff = (lane & 16) ? 16 : 0;     // A-fragment K sub-offset (contiguous k = 16h + e, same order as k_build_b)
    int NT   = H >> 4;                   // 16 N-tiles

    v8f c[4] = {};

    int ksteps = NA >> 5;                // 96
    int ldr = t >> 3;                    // cooperative A load: 16 rows x 8 v4f
    int ldc = (t & 7) * 4;
    const float* arow = newall + (unsigned)(nid[ldr] * NA) + ldc;  // 32-bit index math
    const __bf16* bbase_p = bws + ((unsigned)((wave * 4) * 32 + lane) << 4);
    const unsigned bstep = (unsigned)(NT * 32) << 4;               // elems per K-step

    // prologue: stage A tile 0, preload B fragments of step 0
    {
        v4f a0 = *(const v4f*)(arow);
        *(v4f*)&lds_a[0][ldr * 36 + ldc] = a0;
    }
    v16bf bf0 = *(const v16bf*)(bbase_p);
    v16bf bf1 = *(const v16bf*)(bbase_p + 1 * 32 * 16);
    v16bf bf2 = *(const v16bf*)(bbase_p + 2 * 32 * 16);
    v16bf bf3 = *(const v16bf*)(bbase_p + 3 * 32 * 16);

    for (int s = 0; s < ksteps; ++s) {
        __syncthreads();                 // buf[s&1] stores visible to all waves
        bool hasNext = (s + 1 < ksteps);

        // (1) issue next A tile global load FIRST (in-order returns: its wait
        //     later won't drain the B loads issued after it)
        v4f anext;
        if (hasNext) anext = *(const v4f*)(arow + (s + 1) * 32);

        // (2) build bf16 A fragment from LDS
        //     (lane<16: K{0..15}; lane>=16: K{16..31})
        const float* ab = &lds_a[s & 1][m * 36 + koff];
        v4f q0 = *(const v4f*)(ab);
        v4f q1 = *(const v4f*)(ab + 4);
        v4f q2 = *(const v4f*)(ab + 8);
        v4f q3 = *(const v4f*)(ab + 12);
        v16bf a;
        a[0]  = (__bf16)q0.x; a[1]  = (__bf16)q0.y; a[2]  = (__bf16)q0.z; a[3]  = (__bf16)q0.w;
        a[4]  = (__bf16)q1.x; a[5]  = (__bf16)q1.y; a[6]  = (__bf16)q1.z; a[7]  = (__bf16)q1.w;
        a[8]  = (__bf16)q2.x; a[9]  = (__bf16)q2.y; a[10] = (__bf16)q2.z; a[11] = (__bf16)q2.w;
        a[12] = (__bf16)q3.x; a[13] = (__bf16)q3.y; a[14] = (__bf16)q3.z; a[15] = (__bf16)q3.w;

        // (3) issue B fragment loads for step s+1 (consumed next iteration)
        v16bf nf0, nf1, nf2, nf3;
        if (hasNext) {
            const __bf16* bp = bbase_p + (unsigned)(s + 1) * bstep;
            nf0 = *(const v16bf*)(bp);
            nf1 = *(const v16bf*)(bp + 1 * 32 * 16);
            nf2 = *(const v16bf*)(bp + 2 * 32 * 16);
            nf3 = *(const v16bf*)(bp + 3 * 32 * 16);
        }

        // (4) compute with current fragments
        c[0] = __builtin_amdgcn_wmma_f32_16x16x32_bf16(false, a, false, bf0, (short)0, c[0], false, false);
        c[1] = __builtin_amdgcn_wmma_f32_16x16x32_bf16(false, a, false, bf1, (short)0, c[1], false, false);
        c[2] = __builtin_amdgcn_wmma_f32_16x16x32_bf16(false, a, false, bf2, (short)0, c[2], false, false);
        c[3] = __builtin_amdgcn_wmma_f32_16x16x32_bf16(false, a, false, bf3, (short)0, c[3], false, false);

        // (5) stage next A tile into the other buffer AFTER compute:
        //     the load's wait lands here, hidden behind the 4 WMMAs
        if (hasNext) {
            *(v4f*)&lds_a[(s + 1) & 1][ldr * 36 + ldc] = anext;
            bf0 = nf0; bf1 = nf1; bf2 = nf2; bf3 = nf3;
        }
    }

    // D layout: VGPR r, lane l -> M = r + (l>=16 ? 8 : 0), N = l&15
    int mhi = (lane & 16) ? 8 : 0;
    int ncol0 = wave * 64 + (lane & 15);
    #pragma unroll
    for (int j = 0; j < 4; ++j) {
        int ncol = ncol0 + j * 16;
        #pragma unroll
        for (int r = 0; r < 8; ++r) {
            ofeat[(unsigned)(nid[r + mhi] * H) + ncol] = c[j][r];
        }
    }
}

// ---------------------------------------------------------------------------
// host launcher
// ---------------------------------------------------------------------------
extern "C" void kernel_launch(void* const* d_in, const int* in_sizes, int n_in,
                              void* d_out, int out_size, void* d_ws, size_t ws_size,
                              hipStream_t stream) {
    (void)n_in; (void)out_size; (void)ws_size;
    const float* feat     = (const float*)d_in[0];
    const float* img      = (const float*)d_in[1];
    const float* kg       = (const float*)d_in[2];
    const float* allm     = (const float*)d_in[3];
    const float* img_feat = (const float*)d_in[4];
    const float* kg_feat  = (const float*)d_in[5];
    const float* w1       = (const float*)d_in[6];
    const float* b1       = (const float*)d_in[7];
    const float* w2       = (const float*)d_in[8];
    const float* b2       = (const float*)d_in[9];
    const int*   node_id  = (const int*)d_in[10];
    const int*   pre_id   = (const int*)d_in[11];

    int MIDc = in_sizes[7];                       // 128
    int H    = in_sizes[6] / MIDc;                // 256
    long long N = (long long)in_sizes[0] / H;     // 40000
    int NI = (int)(in_sizes[1] / N);              // 1536
    int NK = (int)(in_sizes[2] / N);              // 1536
    int NA = NI + NK;                             // 3072
    int Bn = in_sizes[10];                        // 20000

    float* ofeat = (float*)d_out;
    float* oimg  = ofeat + (size_t)N * H;
    float* okg   = oimg  + (size_t)N * NI;
    float* oall  = okg   + (size_t)N * NK;

    char* ws = (char*)d_ws;
    int*    flags = (int*)ws;
    size_t  off   = ((size_t)N * sizeof(int) + 255) & ~(size_t)255;
    float*  gate  = (float*)(ws + off);
    off          += ((size_t)Bn * sizeof(float) + 255) & ~(size_t)255;
    __bf16* bws   = (__bf16*)(ws + off);          // NA*H bf16 = 1.5 MB, L2-resident

    k_init_flags<<<(int)((N + 255) / 256), 256, 0, stream>>>(flags, (int)N);
    k_mark_flags<<<(Bn + 255) / 256, 256, 0, stream>>>(node_id, flags, Bn);
    k_copy_rows <<<(int)N, 256, 0, stream>>>(feat, img, kg, allm,
                                             ofeat, oimg, okg, oall, flags, H, NI, NK);
    {
        long long total = (long long)NA * H;
        k_build_b<<<(int)((total + 255) / 256), 256, 0, stream>>>(img_feat, kg_feat, bws, H, NI, NA);
    }
    k_gate<<<Bn, MIDc, 0, stream>>>(feat, node_id, w1, b1, w2, b2, gate, H, MIDc);
    k_blend_softmax<<<Bn, 256, 0, stream>>>(img, kg, node_id, pre_id, gate,
                                            oimg, okg, oall, NI, NK);
    k_attn_wmma<<<(Bn + 15) / 16, 128, 0, stream>>>(oall, bws, node_id, ofeat, Bn, H, NA);
}
